// EdgeLFPEEncoder_14568529068150
// MI455X (gfx1250) — matrix operand order repaired and checked by reading, not compiled, not run on hardware
//
#include <hip/hip_runtime.h>
#include <hip/hip_bf16.h>
#include <math.h>

typedef __attribute__((ext_vector_type(16))) _Float16     v16h;
typedef __attribute__((ext_vector_type(8)))  float        v8f;
typedef __attribute__((ext_vector_type(8)))  unsigned int v8u;

#define WAVES_PER_BLOCK 8
#define THREADS (WAVES_PER_BLOCK * 32)

// ---------------------------------------------------------------------------
// Prep kernel: fold layer2 + fuse into one GEMM.
//   Wc[n][k] = sum_j Wf[n][j] * W2[j][k]   (64 x 32)
//   bc[n]    = bf[n] + sum_j Wf[n][j] * b2[j]
// ---------------------------------------------------------------------------
__global__ void prep_combined(const float* __restrict__ W2,  // (64,32)
                              const float* __restrict__ b2,  // (64,)
                              const float* __restrict__ Wf,  // (64,64)
                              const float* __restrict__ bf,  // (64,)
                              float* __restrict__ Wc,        // (64,32)
                              float* __restrict__ bc)        // (64,)
{
    const int idx = blockIdx.x * blockDim.x + threadIdx.x;
    if (idx < 64 * 32) {
        const int n = idx >> 5, k = idx & 31;
        float s = 0.0f;
#pragma unroll
        for (int j = 0; j < 64; ++j) s += Wf[n * 64 + j] * W2[j * 32 + k];
        Wc[idx] = s;
    }
    if (idx < 64) {
        float s = bf[idx];
#pragma unroll
        for (int j = 0; j < 64; ++j) s += Wf[idx * 64 + j] * b2[j];
        bc[idx] = s;
    }
}

// ---------------------------------------------------------------------------
// B fragment (32x16 f16) for X @ W^T:  B[k][n] = scale * W[ntile*16+n][kchunk*32+k]
// CDNA5 wave32 B layout: lane%16 indexes N, elements index K (table order).
// ---------------------------------------------------------------------------
__device__ __forceinline__ v16h load_bfrag(const float* __restrict__ W, int K,
                                           int ntile, int kchunk, int lane, float scale) {
    v16h b;
    const int k = kchunk * 32 + ((lane >> 4) << 3);
    const float* p = W + (size_t)(ntile * 16 + (lane & 15)) * K + k;
#pragma unroll
    for (int s = 0; s < 16; ++s)
        b[s] = (_Float16)(p[s + 8 * (s >> 3)] * scale);
    return b;
}

__device__ __forceinline__ v8f wmma16(v16h a, v16h b, v8f c) {
    return __builtin_amdgcn_wmma_f32_16x16x32_f16(false, a, false, b,
                                                  (short)0, c, false, false);
}

// Branch-free GELU via CDNA5 hardware v_tanh_f32 (tanh approximation).
// v_nop inside the asm covers the TRANS-op result hazard.
__device__ __forceinline__ float gelu_fast(float x) {
    const float x2 = x * x;
    const float inner = x * __builtin_fmaf(0.0356774081f, x2, 0.7978845608f);
    float t;
    asm("v_tanh_f32 %0, %1\n\tv_nop" : "=v"(t) : "v"(inner));
    return 0.5f * x * (1.0f + t);
}

__global__ void __launch_bounds__(THREADS)
edge_lfpe_wmma(const float* __restrict__ edge_attr,   // (E,64)
               const float* __restrict__ coords,      // (N,3)
               const long long* __restrict__ eidx,    // (2,E) int64
               const float* __restrict__ Wr,          // (32,7)
               const float* __restrict__ W1,          // (32,64)
               const float* __restrict__ b1,          // (32,)
               const float* __restrict__ Wc,          // (64,32) combined
               const float* __restrict__ bc,          // (64,)   combined
               float* __restrict__ out,               // (E,64)
               int E)
{
    __shared__ float sWr[32 * 7];
    __shared__ __align__(16) _Float16 sH[WAVES_PER_BLOCK][16 * 32];  // h staging

    const int tid  = threadIdx.x;
    const int lane = tid & 31;
    const int wib  = tid >> 5;

    for (int i = tid; i < 32 * 7; i += THREADS) sWr[i] = Wr[i];
    __syncthreads();

    const int ncol  = lane & 15;               // D-layout column / B column
    const int mrow  = lane & 15;               // A-layout row (edge within tile)
    const int drow  = (lane >> 4) * 8;         // D-layout row offset (0 or 8)
    const int kbase = (lane < 16) ? 0 : 8;     // A-layout K base

    // ---- loop-invariant weight fragments; 1/sqrt(64) folded into W1 ----
    v16h Bw1[4];   // [c*2+t]  W1 (32x64): K=64 (2 chunks), N=32 (2 tiles)
#pragma unroll
    for (int c = 0; c < 2; ++c)
#pragma unroll
        for (int t = 0; t < 2; ++t)
            Bw1[c * 2 + t] = load_bfrag(W1, 64, t, c, lane, 0.125f);

    v16h Bwc[4];   // Wc (64x32): K=32 (1 chunk), N=64 (4 tiles)
#pragma unroll
    for (int t = 0; t < 4; ++t)
        Bwc[t] = load_bfrag(Wc, 32, t, 0, lane, 1.0f);

    float bias1[2], biasc[4];
#pragma unroll
    for (int t = 0; t < 2; ++t) bias1[t] = b1[t * 16 + ncol];
#pragma unroll
    for (int t = 0; t < 4; ++t) biasc[t] = bc[t * 16 + ncol];

    const int ntiles  = (E + 15) >> 4;
    const int gwave   = blockIdx.x * WAVES_PER_BLOCK + wib;
    const int wstride = gridDim.x * WAVES_PER_BLOCK;

    _Float16* hS = sH[wib];
    const unsigned int* hS32 = (const unsigned int*)hS;

    for (int tile = gwave; tile < ntiles; tile += wstride) {
        const int ebase = tile << 4;

        // prefetch next tile's edge_attr slab (4KB, 128B per lane)
        const int tnext = tile + wstride;
        if (tnext < ntiles)
            __builtin_prefetch(edge_attr + ((size_t)(tnext << 4) * 64 + lane * 32), 0, 0);

        // ---------- geometry + learnable Fourier features ----------
        int e = ebase + mrow; if (e >= E) e = E - 1;
        const long long u = eidx[e];
        const long long v = eidx[(size_t)E + e];
        const float cu0 = coords[u * 3 + 0], cu1 = coords[u * 3 + 1], cu2 = coords[u * 3 + 2];
        const float cv0 = coords[v * 3 + 0], cv1 = coords[v * 3 + 1], cv2 = coords[v * 3 + 2];
        float geom[7];
        geom[0] = cv0 - cu0; geom[1] = cv1 - cu1; geom[2] = cv2 - cu2;
        geom[3] = 0.5f * (cu0 + cv0); geom[4] = 0.5f * (cu1 + cv1); geom[5] = 0.5f * (cu2 + cv2);
        geom[6] = __builtin_amdgcn_sqrtf(geom[0]*geom[0] + geom[1]*geom[1] + geom[2]*geom[2]);

        // A0 = cos half (fourier k 0..31), A1 = sin half (k 32..63); scale is in Bw1
        v16h A0, A1;
#pragma unroll
        for (int i = 0; i < 16; ++i) {
            const int k = kbase + i + 8 * (i >> 3);     // A-layout slot -> K index
            const float* wr = &sWr[k * 7];
            float p = geom[0]*wr[0] + geom[1]*wr[1] + geom[2]*wr[2] + geom[3]*wr[3]
                    + geom[4]*wr[4] + geom[5]*wr[5] + geom[6]*wr[6];
            A0[i] = (_Float16)__cosf(p);
            A1[i] = (_Float16)__sinf(p);
        }

        // ---------- layer 1: h = gelu(fourier @ W1^T + b1) ----------
        v8f hacc[2];
#pragma unroll
        for (int t = 0; t < 2; ++t)
#pragma unroll
            for (int r = 0; r < 8; ++r) hacc[t][r] = bias1[t];
        hacc[0] = wmma16(A0, Bw1[0], hacc[0]);
        hacc[0] = wmma16(A1, Bw1[2], hacc[0]);
        hacc[1] = wmma16(A0, Bw1[1], hacc[1]);
        hacc[1] = wmma16(A1, Bw1[3], hacc[1]);

#pragma unroll
        for (int t = 0; t < 2; ++t)
#pragma unroll
            for (int r = 0; r < 8; ++r)
                hS[(drow + r) * 32 + t * 16 + ncol] = (_Float16)gelu_fast(hacc[t][r]);
        asm volatile("s_wait_dscnt 0" ::: "memory");   // cross-lane staging visible

        // reload h as A fragment (16x32) — two ds_load_b128 per lane
        v8u araw;
        const int abase = mrow * 16 + (kbase >> 1);
#pragma unroll
        for (int j = 0; j < 4; ++j) araw[j] = hS32[abase + j];          // k = kbase+0..7
#pragma unroll
        for (int j = 4; j < 8; ++j) araw[j] = hS32[abase + 4 + j];      // k = kbase+16..23
        const v16h Ah = __builtin_bit_cast(v16h, araw);

        // ---------- combined layer: out = edge_attr + h @ Wc^T + bc ----------
        v8f oacc[4];
#pragma unroll
        for (int t = 0; t < 4; ++t) {
#pragma unroll
            for (int r = 0; r < 8; ++r) oacc[t][r] = biasc[t];
            oacc[t] = wmma16(Ah, Bwc[t], oacc[t]);
        }

#pragma unroll
        for (int t = 0; t < 4; ++t)
#pragma unroll
            for (int r = 0; r < 8; ++r) {
                const int er = ebase + drow + r;
                if (er < E) {
                    const size_t off = (size_t)er * 64 + t * 16 + ncol;
                    out[off] = edge_attr[off] + oacc[t][r];
                }
            }
    }
}

extern "C" void kernel_launch(void* const* d_in, const int* in_sizes, int n_in,
                              void* d_out, int out_size, void* d_ws, size_t ws_size,
                              hipStream_t stream) {
    const float*     edge_attr = (const float*)d_in[0];
    const float*     coords    = (const float*)d_in[1];
    const long long* eidx      = (const long long*)d_in[2];
    const float*     Wr        = (const float*)d_in[3];
    const float*     W1        = (const float*)d_in[4];
    const float*     b1        = (const float*)d_in[5];
    const float*     W2        = (const float*)d_in[6];
    const float*     b2        = (const float*)d_in[7];
    const float*     Wf        = (const float*)d_in[8];
    const float*     bf        = (const float*)d_in[9];
    float*           out       = (float*)d_out;

    // workspace: Wc (64x32) then bc (64)
    float* Wc = (float*)d_ws;
    float* bc = Wc + 64 * 32;

    prep_combined<<<8, 256, 0, stream>>>(W2, b2, Wf, bf, Wc, bc);

    const int E = in_sizes[0] / 64;
    const int ntiles = (E + 15) / 16;
    int blocks = (ntiles + WAVES_PER_BLOCK - 1) / WAVES_PER_BLOCK;
    if (blocks > 1536) blocks = 1536;   // grid-stride: amortize weight-fragment setup
    if (blocks < 1) blocks = 1;

    edge_lfpe_wmma<<<blocks, THREADS, 0, stream>>>(
        edge_attr, coords, eidx, Wr, W1, b1, Wc, bc, out, E);
}
